// ModCRTMoE_21406117003384
// MI455X (gfx1250) — compile-verified, matrix operand order checked by reading, not run on hardware
//
#include <hip/hip_runtime.h>
#include <hip/hip_bf16.h>
#include <math.h>

typedef __attribute__((ext_vector_type(16))) __bf16 v16bf;
typedef __attribute__((ext_vector_type(8)))  float  v8f;
typedef __attribute__((ext_vector_type(8)))  unsigned int v8u;

#define BATCH   2048
#define DIM     768
#define HID     3072
#define OUTD    768
#define NEXP    8
#define MAX_TILES 136                 // worst-case sum_e ceil(cnt_e/16)
#define PERM_CAP (MAX_TILES*16)       // 2176 padded rows

// workspace byte offsets
#define W1B_OFF 32768ull
#define W1B_BYTES (8ull*HID*DIM*2ull)            // 37,748,736
#define W2B_OFF (W1B_OFF + W1B_BYTES)
#define W2B_BYTES (8ull*OUTD*HID*2ull)           // 37,748,736
#define XP_OFF  (W2B_OFF + W2B_BYTES)
#define XP_BYTES ((unsigned long long)PERM_CAP*DIM*2ull)
#define HWS_OFF (XP_OFF + XP_BYTES)

__device__ __forceinline__ unsigned short f2bf(float f) {
    unsigned int u = __float_as_uint(f);
    unsigned int r = u + 0x7FFFu + ((u >> 16) & 1u);   // round-to-nearest-even
    return (unsigned short)(r >> 16);
}
__device__ __forceinline__ unsigned int pack2(float lo, float hi) {
    return (unsigned int)f2bf(lo) | ((unsigned int)f2bf(hi) << 16);
}
__device__ __forceinline__ v16bf joinbf(uint4 lo, uint4 hi) {
    v8u u;
    u[0]=lo.x; u[1]=lo.y; u[2]=lo.z; u[3]=lo.w;
    u[4]=hi.x; u[5]=hi.y; u[6]=hi.z; u[7]=hi.w;
    return __builtin_bit_cast(v16bf, u);
}

// ---------------------------------------------------------------- routing ---
__global__ __launch_bounds__(256) void route_kernel(
    const float* __restrict__ x, const float* __restrict__ Wh,
    const float* __restrict__ bh, int* __restrict__ eid)
{
    int wave = threadIdx.x >> 5, lane = threadIdx.x & 31;
    int t = blockIdx.x * 8 + wave;
    if (t >= BATCH) return;

    const float* xr = x + (size_t)t * DIM;
    float a0 = 0.f, a1 = 0.f, a2 = 0.f, a3 = 0.f;
    for (int j = lane; j < DIM; j += 32) {
        float xv = xr[j];
        float4 w = ((const float4*)Wh)[j];
        a0 += xv * w.x; a1 += xv * w.y; a2 += xv * w.z; a3 += xv * w.w;
    }
    for (int o = 16; o > 0; o >>= 1) {
        a0 += __shfl_xor(a0, o, 32);
        a1 += __shfl_xor(a1, o, 32);
        a2 += __shfl_xor(a2, o, 32);
        a3 += __shfl_xor(a3, o, 32);
    }
    if (lane == 0) {
        const int M[4]   = {3, 5, 7, 11};
        float s[4] = {a0 + bh[0], a1 + bh[1], a2 + bh[2], a3 + bh[3]};
        int r[4];
        #pragma unroll
        for (int k = 0; k < 4; ++k) {
            float f = s[k] - floorf(s[k]);                 // mod 1.0, in [0,1)
            int q = (int)floorf(f * (float)M[k] + 0.5f);
            r[k] = q % M[k];
        }
        const int PI[6]  = {0, 0, 0, 1, 1, 2};
        const int PJ[6]  = {1, 2, 3, 2, 3, 3};
        const int M1a[6] = {3, 3, 3, 5, 5, 7};
        const int M2a[6] = {5, 7, 11, 7, 11, 11};
        const int INVa[6]= {2, 5, 4, 3, 9, 8};
        int bestScore = -1, best = 0;
        #pragma unroll
        for (int p = 0; p < 6; ++p) {
            int r1 = r[PI[p]], r2 = r[PJ[p]];
            int m2 = M2a[p];
            int tt = (((r2 - r1) % m2 + m2) % m2) * INVa[p] % m2;
            int mm = M1a[p] * m2;
            int cand = (r1 + tt * M1a[p]) % mm;
            int sc = 0;
            #pragma unroll
            for (int k = 0; k < 4; ++k) sc += (cand % M[k] == r[k]);
            if (sc > bestScore) { bestScore = sc; best = cand; }  // first argmax
        }
        eid[t] = best % NEXP;
    }
}

// --------------------------------------------------- tile build (1 block) ---
__global__ __launch_bounds__(256) void build_tiles_kernel(
    const int* __restrict__ eid, int* __restrict__ perm, int* __restrict__ texp)
{
    __shared__ int cnt[NEXP];
    __shared__ int off[NEXP + 1];
    __shared__ int slot[NEXP];
    int tid = threadIdx.x;
    if (tid < NEXP) { cnt[tid] = 0; slot[tid] = 0; }
    __syncthreads();
    for (int t = tid; t < BATCH; t += blockDim.x) atomicAdd(&cnt[eid[t]], 1);
    __syncthreads();
    if (tid == 0) {
        int o = 0;
        for (int e = 0; e < NEXP; ++e) {
            off[e] = o;
            o += ((cnt[e] + 15) >> 4) << 4;       // pad per-expert range to 16
        }
        off[NEXP] = o;
    }
    __syncthreads();
    for (int i = tid; i < PERM_CAP; i += blockDim.x) perm[i] = -1;
    for (int i = tid; i < MAX_TILES; i += blockDim.x) {
        int e = -1, row = i * 16;
        for (int q = 0; q < NEXP; ++q)
            if (row >= off[q] && row < off[q + 1]) e = q;
        texp[i] = e;                               // past-the-end tiles -> -1
    }
    __syncthreads();                               // perm fully -1 before fill
    for (int t = tid; t < BATCH; t += blockDim.x) {
        int e = eid[t];
        int p = atomicAdd(&slot[e], 1);
        perm[off[e] + p] = t;
    }
}

// ----------------------------------------------- one-shot fp32 -> bf16 ------
__global__ __launch_bounds__(256) void convert_kernel(
    const float* __restrict__ src, unsigned short* __restrict__ dst, int n4)
{
    int i = blockIdx.x * 256 + threadIdx.x;
    if (i >= n4) return;
    float4 v = ((const float4*)src)[i];
    ((uint2*)dst)[i] = make_uint2(pack2(v.x, v.y), pack2(v.z, v.w));
}

// ------------------------------- perm-ordered, padded bf16 copy of x --------
__global__ __launch_bounds__(192) void gather_x_kernel(
    const float* __restrict__ x, const int* __restrict__ perm,
    unsigned short* __restrict__ Xp)
{
    int row = blockIdx.x;                  // padded row, 0..PERM_CAP-1
    int c4  = threadIdx.x;                 // 0..191 (768/4)
    int tok = perm[row];
    float4 v = make_float4(0.f, 0.f, 0.f, 0.f);
    if (tok >= 0) v = ((const float4*)(x + (size_t)tok * DIM))[c4];
    ((uint2*)(Xp + (size_t)row * DIM))[c4] = make_uint2(pack2(v.x, v.y), pack2(v.z, v.w));
}

// --------------------------------------------------------------- GEMM 1 -----
// H[16 x 3072] = gelu( Xp[16 x 768] * W1b[e]^T ); wave -> 16x32 output,
// A loads reused across two B column groups (2 accumulators).
__global__ __launch_bounds__(256) void gemm1_kernel(
    const unsigned short* __restrict__ Xp, const unsigned short* __restrict__ W1b,
    const int* __restrict__ texp, unsigned short* __restrict__ Hws)
{
    int tile = blockIdx.x;
    int e = texp[tile];
    if (e < 0) return;
    int colbase = blockIdx.y * 256;                  // block covers 256 columns

    int tid = threadIdx.x;
    int wave = tid >> 5, lane = tid & 31;
    int col0 = colbase + wave * 32 + (lane & 15);    // first column group
    int col1 = col0 + 16;                            // second column group
    int kb   = (lane >> 4) * 8;                      // K base for this half-wave
    const unsigned short* Arow  = Xp  + (size_t)(tile * 16 + (lane & 15)) * DIM;
    const unsigned short* Brow0 = W1b + ((size_t)e * HID + col0) * DIM;
    const unsigned short* Brow1 = W1b + ((size_t)e * HID + col1) * DIM;

    v8f acc0 = {}, acc1 = {};
    #pragma unroll 2
    for (int k0 = 0; k0 < DIM; k0 += 32) {
        __builtin_prefetch(Brow0 + k0 + 256, 0, 0);         // global_prefetch_b8
        __builtin_prefetch(Brow1 + k0 + 256, 0, 0);
        uint4 qa0 = *(const uint4*)(Arow + k0 + kb);        // K = kb..kb+7
        uint4 qa1 = *(const uint4*)(Arow + k0 + kb + 16);   // K = kb+16..kb+23
        v16bf a = joinbf(qa0, qa1);
        uint4 qb0 = *(const uint4*)(Brow0 + k0 + kb);
        uint4 qb1 = *(const uint4*)(Brow0 + k0 + kb + 16);
        acc0 = __builtin_amdgcn_wmma_f32_16x16x32_bf16(
                   false, a, false, joinbf(qb0, qb1), (short)0, acc0, false, false);
        uint4 qc0 = *(const uint4*)(Brow1 + k0 + kb);
        uint4 qc1 = *(const uint4*)(Brow1 + k0 + kb + 16);
        acc1 = __builtin_amdgcn_wmma_f32_16x16x32_bf16(
                   false, a, false, joinbf(qc0, qc1), (short)0, acc1, false, false);
    }
    // C/D layout: VGPR r -> M = r (+8 for lanes 16..31), N = lane&15
    int rbase = (lane >> 4) << 3;
    #pragma unroll
    for (int r = 0; r < 8; ++r) {
        size_t rowoff = (size_t)(tile * 16 + rbase + r) * HID;
        float v0 = acc0[r];
        float v1 = acc1[r];
        float g0 = 0.5f * v0 * (1.0f + erff(v0 * 0.70710678118654752f)); // exact gelu
        float g1 = 0.5f * v1 * (1.0f + erff(v1 * 0.70710678118654752f));
        Hws[rowoff + col0] = f2bf(g0);
        Hws[rowoff + col1] = f2bf(g1);
    }
}

// --------------------------------------------------------------- GEMM 2 -----
// out[tok, 768] = H[16 x 3072] * W2b[e]^T + b2[e]; wave -> 16x32 output.
__global__ __launch_bounds__(256) void gemm2_kernel(
    const unsigned short* __restrict__ Hws, const unsigned short* __restrict__ W2b,
    const float* __restrict__ b2p, const int* __restrict__ perm,
    const int* __restrict__ texp, float* __restrict__ out)
{
    int tile = blockIdx.x;
    int e = texp[tile];
    if (e < 0) return;
    int colbase = blockIdx.y * 256;

    int tid = threadIdx.x;
    int wave = tid >> 5, lane = tid & 31;
    int col0 = colbase + wave * 32 + (lane & 15);
    int col1 = col0 + 16;
    int kb   = (lane >> 4) * 8;
    const unsigned short* Arow  = Hws + (size_t)(tile * 16 + (lane & 15)) * HID;
    const unsigned short* Brow0 = W2b + ((size_t)e * OUTD + col0) * HID;
    const unsigned short* Brow1 = W2b + ((size_t)e * OUTD + col1) * HID;

    v8f acc0 = {}, acc1 = {};
    #pragma unroll 2
    for (int k0 = 0; k0 < HID; k0 += 32) {
        __builtin_prefetch(Brow0 + k0 + 256, 0, 0);
        __builtin_prefetch(Brow1 + k0 + 256, 0, 0);
        uint4 qa0 = *(const uint4*)(Arow + k0 + kb);
        uint4 qa1 = *(const uint4*)(Arow + k0 + kb + 16);
        v16bf a = joinbf(qa0, qa1);
        uint4 qb0 = *(const uint4*)(Brow0 + k0 + kb);
        uint4 qb1 = *(const uint4*)(Brow0 + k0 + kb + 16);
        acc0 = __builtin_amdgcn_wmma_f32_16x16x32_bf16(
                   false, a, false, joinbf(qb0, qb1), (short)0, acc0, false, false);
        uint4 qc0 = *(const uint4*)(Brow1 + k0 + kb);
        uint4 qc1 = *(const uint4*)(Brow1 + k0 + kb + 16);
        acc1 = __builtin_amdgcn_wmma_f32_16x16x32_bf16(
                   false, a, false, joinbf(qc0, qc1), (short)0, acc1, false, false);
    }
    float bias0 = b2p[(size_t)e * OUTD + col0];
    float bias1 = b2p[(size_t)e * OUTD + col1];
    int rbase = (lane >> 4) << 3;
    #pragma unroll
    for (int r = 0; r < 8; ++r) {
        int row = rbase + r;
        int tok = perm[tile * 16 + row];
        if (tok >= 0) {
            out[(size_t)tok * OUTD + col0] = acc0[r] + bias0;
            out[(size_t)tok * OUTD + col1] = acc1[r] + bias1;
        }
    }
}

// ---------------------------------------------------------------- launch ----
extern "C" void kernel_launch(void* const* d_in, const int* in_sizes, int n_in,
                              void* d_out, int out_size, void* d_ws, size_t ws_size,
                              hipStream_t stream) {
    const float* x  = (const float*)d_in[0];
    const float* Wh = (const float*)d_in[1];
    const float* bh = (const float*)d_in[2];
    const float* W1 = (const float*)d_in[3];
    const float* W2 = (const float*)d_in[4];
    const float* b2 = (const float*)d_in[5];
    float* out = (float*)d_out;

    int* eid  = (int*)d_ws;
    int* perm = eid + BATCH;
    int* texp = perm + PERM_CAP;
    unsigned short* W1b = (unsigned short*)((char*)d_ws + W1B_OFF);
    unsigned short* W2b = (unsigned short*)((char*)d_ws + W2B_OFF);
    unsigned short* Xp  = (unsigned short*)((char*)d_ws + XP_OFF);
    unsigned short* Hws = (unsigned short*)((char*)d_ws + HWS_OFF);

    route_kernel<<<BATCH / 8, 256, 0, stream>>>(x, Wh, bh, eid);
    build_tiles_kernel<<<1, 256, 0, stream>>>(eid, perm, texp);

    int n4w1 = NEXP * HID * DIM / 4;       // 4,718,592 float4s
    int n4w2 = NEXP * OUTD * HID / 4;
    convert_kernel<<<(n4w1 + 255) / 256, 256, 0, stream>>>(W1, W1b, n4w1);
    convert_kernel<<<(n4w2 + 255) / 256, 256, 0, stream>>>(W2, W2b, n4w2);
    gather_x_kernel<<<PERM_CAP, 192, 0, stream>>>(x, perm, Xp);

    dim3 g1(MAX_TILES, HID / 256);         // block covers 256 hidden columns
    gemm1_kernel<<<g1, 256, 0, stream>>>(Xp, W1b, texp, Hws);
    dim3 g2(MAX_TILES, OUTD / 256);
    gemm2_kernel<<<g2, 256, 0, stream>>>(Hws, W2b, b2, perm, texp, out);
}
